// Net_76321568849923
// MI455X (gfx1250) — matrix operand order repaired and checked by reading, not compiled, not run on hardware
//
#include <hip/hip_runtime.h>
#include <hip/hip_bf16.h>

#define DIM     32
#define NFEAT   11
#define NNODES  65536
#define NEDGES  262144

typedef __attribute__((ext_vector_type(16))) __bf16 v16bf;
typedef __attribute__((ext_vector_type(8)))  float  v8f;

// ---- workspace layout (bytes, 256B-aligned blocks) ----
// Wt  : [E][32 o][32 i] bf16  (per-edge weight matrices, TRANSPOSED: column-contiguous)
#define OFF_W    0ul
// w2p : packed bf16 nn_w2 in WMMA-B fragment order [64 ntile][2 slice][32 lane][16 t]
#define OFF_W2P  536870912ul
// h   : [N][32] f32
#define OFF_H    537001984ul
// agg : [N][32] f32
#define OFF_AGG  545390592ul
// counts: [N] f32
#define OFF_CNT  553779200ul
// g   : [32] f32 column sums
#define OFF_G    554041344ul

// h = relu(x @ lin0_w + lin0_b); one wave per node, lane = out channel
__global__ void k_lin0(const float* __restrict__ x, const float* __restrict__ w,
                       const float* __restrict__ b, float* __restrict__ h) {
    int node = (blockIdx.x * blockDim.x + threadIdx.x) >> 5;
    int lane = threadIdx.x & 31;
    if (node >= NNODES) return;
    float acc = b[lane];
    #pragma unroll
    for (int f = 0; f < NFEAT; ++f)
        acc += x[node * NFEAT + f] * w[f * DIM + lane];
    h[node * DIM + lane] = fmaxf(acc, 0.0f);
}

// in-degree counts via f32 atomics
__global__ void k_count(const int* __restrict__ ei, float* __restrict__ counts) {
    int e = blockIdx.x * blockDim.x + threadIdx.x;
    if (e >= NEDGES) return;
    int dst = ei[NEDGES + e];
    __hip_atomic_fetch_add(&counts[dst], 1.0f, __ATOMIC_RELAXED, __HIP_MEMORY_SCOPE_AGENT);
}

// Pre-pack nn_w2 (fp32 [64][1024]) to bf16 in WMMA B-fragment order:
// w2p[((ntile*2+s)*32 + lane)*16 + t] = w2[((t&7) + 8*(lane>>4) + 16*(t>>3) + 32*s)*1024 + ntile*16 + (lane&15)]
__global__ void k_packw2(const float* __restrict__ w2, __bf16* __restrict__ w2p) {
    int u = blockIdx.x * blockDim.x + threadIdx.x;   // 65536 elements
    if (u >= 64 * 1024) return;
    int t  = u & 15;
    int L  = (u >> 4) & 31;
    int s  = (u >> 9) & 1;
    int nt = u >> 10;
    int K  = (t & 7) + 8 * (L >> 4) + 16 * (t >> 3) + 32 * s;
    int n  = nt * 16 + (L & 15);
    w2p[u] = (__bf16)w2[K * 1024 + n];
}

// Edge-weight GEMM via WMMA: ew[e] = relu(ea[e]*w1 + b1) @ nn_w2 + b2.
// B (128KB) is staged once per block into LDS with async global->LDS DMA,
// then every wave feeds its WMMAs from ds_load instead of re-streaming L2.
// Output stored bf16, transposed per edge: Wt[e][o][i] = ew[e][i*32+o].
__global__ void k_edgew(const float* __restrict__ ea, const float* __restrict__ w1,
                        const float* __restrict__ b1, const float* __restrict__ b2,
                        const __bf16* __restrict__ w2p, __bf16* __restrict__ Wt) {
    extern __shared__ char smem[];   // 131072 bytes: LDS copy of w2p

    // --- async stage: 256 threads x 512B each, 16B per async op ---
    {
        int t = threadIdx.x;
        const char* gsrc = (const char*)w2p + (size_t)t * 512;
        unsigned    loff = (unsigned)(uintptr_t)(smem + t * 512);
        #pragma unroll 4
        for (int i = 0; i < 32; ++i) {
            unsigned    lva = loff + i * 16;
            const char* ga  = gsrc + i * 16;
            asm volatile("global_load_async_to_lds_b128 %0, %1, off"
                         :: "v"(lva), "v"(ga) : "memory");
        }
        asm volatile("s_wait_asynccnt 0x0" ::: "memory");
    }
    __syncthreads();

    int mtile = (blockIdx.x * blockDim.x + threadIdx.x) >> 5;
    int lane  = threadIdx.x & 31;
    if (mtile >= NEDGES / 16) return;
    int m = lane & 15, hi = lane >> 4;

    // A fragment (16x64 hidden relu activations) computed directly in the
    // documented 16-bit A VGPR layout: lane L holds K = (t&7)+8*hi+16*(t>>3) (+32*slice)
    float a = ea[(long)mtile * 16 + m];
    v16bf A0, A1;
    #pragma unroll
    for (int t = 0; t < 16; ++t) {
        int K = (t & 7) + 8 * hi + 16 * (t >> 3);
        A0[t] = (__bf16)fmaxf(a * w1[K]      + b1[K],      0.0f);
        A1[t] = (__bf16)fmaxf(a * w1[K + 32] + b1[K + 32], 0.0f);
    }

    for (int nt = 0; nt < 64; ++nt) {
        int j = nt * 16 + m;              // flattened (i,o) column of this tile
        float cv = b2[j];
        v8f c = {cv, cv, cv, cv, cv, cv, cv, cv};
        // B fragments from LDS: fragment = 32 contiguous bytes per lane
        v16bf B0 = *(const v16bf*)(smem + ((size_t)(nt * 2 + 0) * 32 + lane) * 32);
        v16bf B1 = *(const v16bf*)(smem + ((size_t)(nt * 2 + 1) * 32 + lane) * 32);
        c = __builtin_amdgcn_wmma_f32_16x16x32_bf16(false, A0, false, B0, (short)0, c, false, false);
        c = __builtin_amdgcn_wmma_f32_16x16x32_bf16(false, A1, false, B1, (short)0, c, false, false);
        int i_idx = j >> 5;               // input row of W[e]
        int o_idx = j & 31;               // output col of W[e]
        #pragma unroll
        for (int r = 0; r < 8; ++r) {     // D row M = r + 8*hi = edge within tile
            long e2 = (long)mtile * 16 + r + 8 * hi;
            Wt[e2 * 1024 + o_idx * 32 + i_idx] = (__bf16)c[r];
        }
    }
}

// msg[e] = h[src[e]] @ W[e]; one wave per edge, lane = output channel.
// Wt layout gives each lane its column as 64 contiguous bytes -> 4x b128 NT loads.
// src/dst are wave-uniform -> h row read through a scalar-uniform pointer (no shuffles).
__global__ void k_msg(const int* __restrict__ ei, const float* __restrict__ h,
                      const __bf16* __restrict__ Wt, float* __restrict__ agg) {
    int wid  = (blockIdx.x * blockDim.x + threadIdx.x) >> 5;
    int lane = threadIdx.x & 31;
    if (wid >= NEDGES) return;
    int e   = __builtin_amdgcn_readfirstlane(wid);
    int src = __builtin_amdgcn_readfirstlane(ei[e]);
    int dst = __builtin_amdgcn_readfirstlane(ei[NEDGES + e]);

    const v16bf* wp = (const v16bf*)(Wt + (size_t)e * 1024 + lane * 32);
    v16bf w0 = __builtin_nontemporal_load(wp);
    v16bf w1 = __builtin_nontemporal_load(wp + 1);

    const float* hr = h + (size_t)src * DIM;   // uniform address
    float acc = 0.0f;
    #pragma unroll
    for (int i = 0; i < 16; ++i) acc += hr[i]      * (float)w0[i];
    #pragma unroll
    for (int i = 0; i < 16; ++i) acc += hr[16 + i] * (float)w1[i];

    __hip_atomic_fetch_add(&agg[(size_t)dst * DIM + lane], acc,
                           __ATOMIC_RELAXED, __HIP_MEMORY_SCOPE_AGENT);
}

// h += relu(agg/deg + h@root + bias); re-zeroes agg for the next iteration.
// h row is wave-uniform -> uniform reads for the root matvec (no shuffles).
__global__ void k_update(float* __restrict__ h, float* __restrict__ agg,
                         const float* __restrict__ counts, const float* __restrict__ root,
                         const float* __restrict__ cbias) {
    int wid  = (blockIdx.x * blockDim.x + threadIdx.x) >> 5;
    int lane = threadIdx.x & 31;
    if (wid >= NNODES) return;
    int n = __builtin_amdgcn_readfirstlane(wid);

    const float* hr = h + (size_t)n * DIM;     // uniform address
    float hv = h[(size_t)n * DIM + lane];      // lane's own value (vector load)
    float rp = 0.0f;
    #pragma unroll
    for (int i = 0; i < DIM; ++i)
        rp += hr[i] * root[i * DIM + lane];

    float av = agg[(size_t)n * DIM + lane];
    agg[(size_t)n * DIM + lane] = 0.0f;
    float cnt = counts[n];
    float inv = cnt > 0.0f ? 1.0f / cnt : 0.0f;
    float u = av * inv + rp + cbias[lane];
    h[(size_t)n * DIM + lane] = hv + fmaxf(u, 0.0f);
}

// column sums of h into g[32]
__global__ void k_colsum(const float* __restrict__ h, float* __restrict__ g) {
    int lane  = threadIdx.x & 31;
    int row0  = (blockIdx.x * blockDim.x + threadIdx.x) >> 5;
    int strd  = (gridDim.x * blockDim.x) >> 5;
    float acc = 0.0f;
    for (int n = row0; n < NNODES; n += strd)
        acc += h[n * DIM + lane];
    __hip_atomic_fetch_add(&g[lane], acc, __ATOMIC_RELAXED, __HIP_MEMORY_SCOPE_AGENT);
}

// out = (g/N) @ lin2_w + lin2_b ; single wave
__global__ void k_final(const float* __restrict__ g, const float* __restrict__ w,
                        const float* __restrict__ b, float* __restrict__ out) {
    int lane = threadIdx.x;
    float v = (g[lane] * (1.0f / (float)NNODES)) * w[lane];
    #pragma unroll
    for (int off = 16; off > 0; off >>= 1)
        v += __shfl_down(v, off, 32);
    if (lane == 0) out[0] = v + b[0];
}

extern "C" void kernel_launch(void* const* d_in, const int* in_sizes, int n_in,
                              void* d_out, int out_size, void* d_ws, size_t ws_size,
                              hipStream_t stream) {
    (void)in_sizes; (void)n_in; (void)out_size; (void)ws_size;
    const float* x      = (const float*)d_in[0];
    const int*   ei     = (const int*)  d_in[1];
    const float* ea     = (const float*)d_in[2];
    const float* lin0_w = (const float*)d_in[3];
    const float* lin0_b = (const float*)d_in[4];
    const float* nn_w1  = (const float*)d_in[5];
    const float* nn_b1  = (const float*)d_in[6];
    const float* nn_w2  = (const float*)d_in[7];
    const float* nn_b2  = (const float*)d_in[8];
    const float* root   = (const float*)d_in[9];
    const float* cbias  = (const float*)d_in[10];
    const float* lin2_w = (const float*)d_in[11];
    const float* lin2_b = (const float*)d_in[12];

    char* ws = (char*)d_ws;
    __bf16* Wt     = (__bf16*)(ws + OFF_W);
    __bf16* w2p    = (__bf16*)(ws + OFF_W2P);
    float*  h      = (float*) (ws + OFF_H);
    float*  agg    = (float*) (ws + OFF_AGG);
    float*  counts = (float*) (ws + OFF_CNT);
    float*  g      = (float*) (ws + OFF_G);

    hipMemsetAsync(agg,    0, (size_t)NNODES * DIM * sizeof(float), stream);
    hipMemsetAsync(counts, 0, (size_t)NNODES * sizeof(float),       stream);
    hipMemsetAsync(g,      0, (size_t)DIM * sizeof(float),          stream);

    k_lin0  <<<NNODES / 8,        256, 0, stream>>>(x, lin0_w, lin0_b, h);
    k_count <<<NEDGES / 256,      256, 0, stream>>>(ei, counts);
    k_packw2<<<(64 * 1024) / 256, 256, 0, stream>>>(nn_w2, w2p);
    k_edgew <<<(NEDGES / 16) / 8, 256, 131072, stream>>>(ea, nn_w1, nn_b1, nn_b2, w2p, Wt);

    for (int it = 0; it < 8; ++it) {
        k_msg   <<<NEDGES / 8, 256, 0, stream>>>(ei, h, Wt, agg);
        k_update<<<NNODES / 8, 256, 0, stream>>>(h, agg, counts, root, cbias);
    }

    k_colsum<<<512, 256, 0, stream>>>(h, g);
    k_final <<<1,    32, 0, stream>>>(g, lin2_w, lin2_b, (float*)d_out);
}
